// SourceEncoder_52355651338901
// MI455X (gfx1250) — matrix operand order repaired and checked by reading, not compiled, not run on hardware
//
#include <hip/hip_runtime.h>
#include <hip/hip_bf16.h>

// ---------------- types for WMMA ----------------
typedef __attribute__((ext_vector_type(16))) _Float16 v16h;
typedef __attribute__((ext_vector_type(8)))  _Float16 v8h;
typedef __attribute__((ext_vector_type(8)))  float    v8f;

union V16U { v16h v; v8h h[2]; };

// A fragment (16x32 f16, row-major, leading dim K), pointer pre-offset by
// (lane<16?0:16): load halves at +0 and +8.
__device__ __forceinline__ v16h loadA16(const _Float16* p) {
  V16U u;
  u.h[0] = *(const v8h*)(p);
  u.h[1] = *(const v8h*)(p + 8);
  return u.v;
}
// B fragment (32x16 f16): 16 contiguous f16 of one W row (W row-major NxK),
// pointer pre-offset by (lane<16?0:16).
__device__ __forceinline__ v16h loadB16(const _Float16* p) {
  V16U u;
  u.h[0] = *(const v8h*)(p);
  u.h[1] = *(const v8h*)(p + 8);
  return u.v;
}

#define WMMA_F16(A, B, C) \
  __builtin_amdgcn_wmma_f32_16x16x32_f16(false, (A), false, (B), (short)0, (C), false, false)

// ---------------- problem constants ----------------
#define SLEN   104
#define NTXY   49                // (104-8)/2+1
#define NT_IMG (NTXY * NTXY)     // 2401
#define NBATCH 32
#define NT_TOT (NBATCH * NT_IMG) // 76832 = 16 * 4802
#define MT16   (NT_TOT / 16)     // 4802
#define FLAT_K 1280
#define HID    256
#define DIMOUT 27
#define C2K    288               // conv2 K: 9 tap-blocks * 32 padded channels
#define H1PSZ  (100 * 32)        // padded h1, pixel-major channel-minor(32)

// ======================================================================
// 1) Weight prep: FC weights fp32 -> f16 (fcf padded to 32x256) AND the
//    conv B matrices in WMMA-friendly layouts (built once, globally):
//    w1col (32x32):  k = i*9+r (18 valid), row n < 20 valid
//    b2col (32x288): k = blk*32+ch, value = conv2_w[n][ch][ky][kx]
// ======================================================================
__global__ void convert_weights_kernel(const float* __restrict__ fc1w,
                                       const float* __restrict__ fc2w,
                                       const float* __restrict__ fc3w,
                                       const float* __restrict__ fcfw,
                                       const float* __restrict__ conv1w,
                                       const float* __restrict__ conv2w,
                                       _Float16* __restrict__ w1h,
                                       _Float16* __restrict__ w2h,
                                       _Float16* __restrict__ w3h,
                                       _Float16* __restrict__ wfh,
                                       _Float16* __restrict__ w1colg,
                                       _Float16* __restrict__ b2colg) {
  int i = blockIdx.x * 256 + threadIdx.x;
  if (i < HID * FLAT_K) w1h[i] = (_Float16)fc1w[i];
  if (i < HID * HID) {
    w2h[i] = (_Float16)fc2w[i];
    w3h[i] = (_Float16)fc3w[i];
  }
  if (i < 32 * HID) {
    int row = i >> 8;
    wfh[i] = (row < DIMOUT) ? (_Float16)fcfw[i] : (_Float16)0.0f;
  }
  if (i < 32 * 32) {
    int n = i >> 5, k = i & 31;
    w1colg[i] = (n < 20 && k < 18) ? (_Float16)conv1w[n * 18 + k] : (_Float16)0.0f;
  }
  if (i < 32 * C2K) {
    int n = i / C2K, k = i - n * C2K;
    int blk = k >> 5, ch = k & 31;
    b2colg[i] = (n < 20 && ch < 20) ? (_Float16)conv2w[n * 180 + ch * 9 + blk]
                                    : (_Float16)0.0f;
  }
}

// ======================================================================
// 2) Fully-WMMA conv pipeline. 4 tiles per WG (1 wave per tile).
//    conv1: 64x32 @ 32x32 (K=18 padded to 32, ONE k-step), A = tiny im2col.
//    conv2: 64x288 @ 288x32, h1 pixel-major/channel-minor(32) so every A
//    fragment is two contiguous ds_load_b128. Branchless epilogues:
//    pad-lane results are exact zeros (zero B rows + selected zero bias);
//    conv2 pad stores go to a dump pointer (v_cndmask, no exec branches).
// ======================================================================
__global__ __launch_bounds__(128) void conv_tiles_kernel(
    const float* __restrict__ img,
    const _Float16* __restrict__ w1colg, const float* __restrict__ b1,
    const _Float16* __restrict__ b2colg, const float* __restrict__ b2,
    _Float16* __restrict__ act0, _Float16* __restrict__ dump) {
  __shared__ __align__(16) _Float16 s_w1col[32 * 32];      // conv1 B (N=32,K=32)
  __shared__ __align__(16) _Float16 s_b2col[32 * C2K];     // conv2 B (N=32,K=288)
  __shared__ __align__(16) _Float16 s_in[4][200];          // padded input (2x10x10)
  __shared__ __align__(16) _Float16 s_a1[4][64 * 32];      // conv1 im2col
  __shared__ __align__(16) _Float16 s_h1p[4][H1PSZ];       // padded h1 [pix][32]

  const int tid = threadIdx.x;
  // vectorized B-matrix fill (global b128 -> LDS b128)
  {
    v8h* d0 = (v8h*)s_w1col;
    const v8h* s0 = (const v8h*)w1colg;
    for (int u = tid; u < (32 * 32) / 8; u += 128) d0[u] = s0[u];
    v8h* d1 = (v8h*)s_b2col;
    const v8h* s1 = (const v8h*)b2colg;
    for (int u = tid; u < (32 * C2K) / 8; u += 128) d1[u] = s1[u];
  }
  // vectorized zero-init (input halos, im2col pad cols, h1p halos+pad ch)
  {
    const v8h z = {};
    v8h* zi = (v8h*)&s_in[0][0];
    for (int u = tid; u < (4 * 200) / 8; u += 128) zi[u] = z;
    v8h* za = (v8h*)&s_a1[0][0];
    for (int u = tid; u < (4 * 64 * 32) / 8; u += 128) za[u] = z;
    v8h* zh = (v8h*)&s_h1p[0][0];
    for (int u = tid; u < (4 * H1PSZ) / 8; u += 128) zh[u] = z;
  }
  __syncthreads();

  const int w    = tid >> 5;
  const int lane = tid & 31;
  const int hs   = lane >> 4;
  const int nl   = lane & 15;
  const int t    = blockIdx.x * 4 + w;  // exact: grid = NT/4
  const int b    = t / NT_IMG;
  const int rem  = t % NT_IMG;
  const int oy   = (rem / NTXY) * 2;
  const int ox   = (rem % NTXY) * 2;

  // load 2x8x8 interior -> f16 padded LDS
  for (int e = lane; e < 128; e += 32) {
    int c = e >> 6, p = e & 63, y = p >> 3, x = p & 7;
    s_in[w][c * 100 + (y + 1) * 10 + (x + 1)] =
        (_Float16)img[(((size_t)b * 2 + c) * SLEN + (oy + y)) * SLEN + (ox + x)];
  }
  __syncthreads();

  // conv1 im2col: 2 pixels per lane, 18 entries each, constant LDS offsets
  {
    const _Float16* sin = &s_in[w][0];
    #pragma unroll
    for (int pp = 0; pp < 2; ++pp) {
      int p = lane * 2 + pp, y = p >> 3, x = p & 7;
      const _Float16* src = sin + y * 10 + x;
      _Float16* row = &s_a1[w][p * 32];
      #pragma unroll
      for (int i = 0; i < 2; ++i)
        #pragma unroll
        for (int ky = 0; ky < 3; ++ky)
          #pragma unroll
          for (int kx = 0; kx < 3; ++kx)
            row[i * 9 + ky * 3 + kx] = src[i * 100 + ky * 10 + kx];
    }
  }
  __syncthreads();

  // ---- conv1 WMMA: single K=32 step, 4 m-tiles x 2 n-tiles ----
  {
    v16h bb0 = loadB16(&s_w1col[nl * 32 + hs * 16]);
    v16h bb1 = loadB16(&s_w1col[(16 + nl) * 32 + hs * 16]);
    const float c0 = b1[nl];
    const float c1t = b1[16 + (nl & 3)];            // always in-bounds
    const float c1  = (nl < 4) ? c1t : 0.0f;        // value select, no branch
    #pragma unroll
    for (int mt = 0; mt < 4; ++mt) {
      v16h a = loadA16(&s_a1[w][(mt * 16 + nl) * 32 + hs * 16]);
      v8f d0 = WMMA_F16(a, bb0, (v8f){});
      v8f d1 = WMMA_F16(a, bb1, (v8f){});
      #pragma unroll
      for (int r = 0; r < 8; ++r) {
        int p = mt * 16 + hs * 8 + r, y = p >> 3, x = p & 7;
        int pix = (y + 1) * 10 + (x + 1);
        s_h1p[w][pix * 32 + nl] = (_Float16)fmaxf(d0[r] + c0, 0.0f);
        // pad channels: d1==0 and c1==0 -> stores exact 0 (zero region stays 0)
        s_h1p[w][pix * 32 + 16 + nl] = (_Float16)fmaxf(d1[r] + c1, 0.0f);
      }
    }
  }
  __syncthreads();

  // ---- conv2 WMMA: 9 tap-block K-steps, A = contiguous h1p slices ----
  const _Float16* h1 = &s_h1p[w][0];
  v8f acc[4][2];
  #pragma unroll
  for (int mt = 0; mt < 4; ++mt) { acc[mt][0] = (v8f){}; acc[mt][1] = (v8f){}; }

  #pragma unroll
  for (int blk = 0; blk < 9; ++blk) {
    const int ky = blk / 3, kx = blk - 3 * ky;
    v16h bb0 = loadB16(&s_b2col[nl * C2K + blk * 32 + hs * 16]);
    v16h bb1 = loadB16(&s_b2col[(16 + nl) * C2K + blk * 32 + hs * 16]);
    #pragma unroll
    for (int mt = 0; mt < 4; ++mt) {
      int p = mt * 16 + nl, y = p >> 3, x = p & 7;
      v16h a = loadA16(&h1[((y + ky) * 10 + (x + kx)) * 32 + hs * 16]);
      acc[mt][0] = WMMA_F16(a, bb0, acc[mt][0]);
      acc[mt][1] = WMMA_F16(a, bb1, acc[mt][1]);
    }
  }

  // epilogue: bias + relu -> flattened f16 act0, packed b128 stores.
  // second n-tile: pointer select (valid channels 16..19 else dump scratch).
  const float bn0 = b2[nl];
  const float bt  = b2[16 + (nl & 3)];
  const float bn1 = (nl < 4) ? bt : 0.0f;
  _Float16* outp = act0 + (size_t)t * FLAT_K;
  _Float16* out1 = (nl < 4) ? (outp + (16 + nl) * 64) : dump;
  #pragma unroll
  for (int mt = 0; mt < 4; ++mt) {
    v8h p0, p1;
    #pragma unroll
    for (int r = 0; r < 8; ++r) {
      p0[r] = (_Float16)fmaxf(acc[mt][0][r] + bn0, 0.0f);
      p1[r] = (_Float16)fmaxf(acc[mt][1][r] + bn1, 0.0f);
    }
    *(v8h*)(outp + nl * 64 + mt * 16 + hs * 8) = p0;
    *(v8h*)(out1 + mt * 16 + hs * 8) = p1;
  }
}

// ======================================================================
// 3) Fused MLP: fc1(1280->256)+relu, fc2, fc3 (LDS-resident 16x256 rows),
//    fcf (256->27) -> fp32 out. Block = 128 thr (4 waves), 16-row m-tile;
//    the 4 waves split the 256 columns so act0 is read once per tile.
// ======================================================================
__global__ __launch_bounds__(128) void mlp_kernel(
    const _Float16* __restrict__ act0,
    const _Float16* __restrict__ w1h, const _Float16* __restrict__ w2h,
    const _Float16* __restrict__ w3h, const _Float16* __restrict__ wfh,
    const float* __restrict__ b1, const float* __restrict__ b2,
    const float* __restrict__ b3, const float* __restrict__ bf,
    float* __restrict__ out) {
  __shared__ __align__(16) _Float16 s_a[16 * HID];
  __shared__ __align__(16) _Float16 s_b[16 * HID];

  const int wave = threadIdx.x >> 5;
  const int lane = threadIdx.x & 31;
  const int hs   = lane >> 4;
  const int nl   = lane & 15;
  const int row0 = blockIdx.x * 16;
  const int col  = wave * 64 + nl;

  // ---- fc1: K=1280, A from global ----
  {
    const _Float16* aptr = act0 + (size_t)(row0 + nl) * FLAT_K + hs * 16;
    const _Float16* bptr = w1h + (size_t)col * FLAT_K + hs * 16;
    const size_t bs = (size_t)16 * FLAT_K;
    v8f a0 = {}, a1 = {}, a2 = {}, a3 = {};
    for (int k = 0; k < FLAT_K; k += 32) {
      v16h a  = loadA16(aptr + k);
      v16h v0 = loadB16(bptr + k);
      v16h v1 = loadB16(bptr + bs + k);
      v16h v2 = loadB16(bptr + 2 * bs + k);
      v16h v3 = loadB16(bptr + 3 * bs + k);
      a0 = WMMA_F16(a, v0, a0); a1 = WMMA_F16(a, v1, a1);
      a2 = WMMA_F16(a, v2, a2); a3 = WMMA_F16(a, v3, a3);
    }
    const float c0 = b1[col], c1 = b1[col + 16], c2 = b1[col + 32], c3 = b1[col + 48];
    #pragma unroll
    for (int r = 0; r < 8; ++r) {
      int m = r + hs * 8;
      s_a[m * HID + col]      = (_Float16)fmaxf(a0[r] + c0, 0.0f);
      s_a[m * HID + col + 16] = (_Float16)fmaxf(a1[r] + c1, 0.0f);
      s_a[m * HID + col + 32] = (_Float16)fmaxf(a2[r] + c2, 0.0f);
      s_a[m * HID + col + 48] = (_Float16)fmaxf(a3[r] + c3, 0.0f);
    }
  }
  __syncthreads();

  // ---- fc2: K=256, A from LDS s_a -> s_b ----
  {
    const _Float16* aptr = s_a + (size_t)nl * HID + hs * 16;
    const _Float16* bptr = w2h + (size_t)col * HID + hs * 16;
    const size_t bs = (size_t)16 * HID;
    v8f a0 = {}, a1 = {}, a2 = {}, a3 = {};
    for (int k = 0; k < HID; k += 32) {
      v16h a  = loadA16(aptr + k);
      v16h v0 = loadB16(bptr + k);
      v16h v1 = loadB16(bptr + bs + k);
      v16h v2 = loadB16(bptr + 2 * bs + k);
      v16h v3 = loadB16(bptr + 3 * bs + k);
      a0 = WMMA_F16(a, v0, a0); a1 = WMMA_F16(a, v1, a1);
      a2 = WMMA_F16(a, v2, a2); a3 = WMMA_F16(a, v3, a3);
    }
    const float c0 = b2[col], c1 = b2[col + 16], c2 = b2[col + 32], c3 = b2[col + 48];
    #pragma unroll
    for (int r = 0; r < 8; ++r) {
      int m = r + hs * 8;
      s_b[m * HID + col]      = (_Float16)fmaxf(a0[r] + c0, 0.0f);
      s_b[m * HID + col + 16] = (_Float16)fmaxf(a1[r] + c1, 0.0f);
      s_b[m * HID + col + 32] = (_Float16)fmaxf(a2[r] + c2, 0.0f);
      s_b[m * HID + col + 48] = (_Float16)fmaxf(a3[r] + c3, 0.0f);
    }
  }
  __syncthreads();

  // ---- fc3: K=256, A from LDS s_b -> s_a ----
  {
    const _Float16* aptr = s_b + (size_t)nl * HID + hs * 16;
    const _Float16* bptr = w3h + (size_t)col * HID + hs * 16;
    const size_t bs = (size_t)16 * HID;
    v8f a0 = {}, a1 = {}, a2 = {}, a3 = {};
    for (int k = 0; k < HID; k += 32) {
      v16h a  = loadA16(aptr + k);
      v16h v0 = loadB16(bptr + k);
      v16h v1 = loadB16(bptr + bs + k);
      v16h v2 = loadB16(bptr + 2 * bs + k);
      v16h v3 = loadB16(bptr + 3 * bs + k);
      a0 = WMMA_F16(a, v0, a0); a1 = WMMA_F16(a, v1, a1);
      a2 = WMMA_F16(a, v2, a2); a3 = WMMA_F16(a, v3, a3);
    }
    const float c0 = b3[col], c1 = b3[col + 16], c2 = b3[col + 32], c3 = b3[col + 48];
    #pragma unroll
    for (int r = 0; r < 8; ++r) {
      int m = r + hs * 8;
      s_a[m * HID + col]      = (_Float16)fmaxf(a0[r] + c0, 0.0f);
      s_a[m * HID + col + 16] = (_Float16)fmaxf(a1[r] + c1, 0.0f);
      s_a[m * HID + col + 32] = (_Float16)fmaxf(a2[r] + c2, 0.0f);
      s_a[m * HID + col + 48] = (_Float16)fmaxf(a3[r] + c3, 0.0f);
    }
  }
  __syncthreads();

  // ---- fcf: wave 0 only, N padded 32, 27 valid cols, fp32 out ----
  if (wave == 0) {
    const _Float16* aptr = s_a + (size_t)nl * HID + hs * 16;
    const _Float16* bptr = wfh + (size_t)nl * HID + hs * 16;
    const size_t bs = (size_t)16 * HID;
    v8f a0 = {}, a1 = {};
    for (int k = 0; k < HID; k += 32) {
      v16h a  = loadA16(aptr + k);
      v16h v0 = loadB16(bptr + k);
      v16h v1 = loadB16(bptr + bs + k);
      a0 = WMMA_F16(a, v0, a0);
      a1 = WMMA_F16(a, v1, a1);
    }
    const float c0 = bf[nl];
    const float c1t = bf[(nl < 11) ? (nl + 16) : 0];  // always in-bounds
    const float c1  = (nl < 11) ? c1t : 0.0f;
    #pragma unroll
    for (int r = 0; r < 8; ++r) {
      size_t rowoff = (size_t)(row0 + r + hs * 8) * DIMOUT;
      out[rowoff + nl] = a0[r] + c0;
      if (nl < 11) out[rowoff + nl + 16] = a1[r] + c1;
    }
  }
}

// ======================================================================
extern "C" void kernel_launch(void* const* d_in, const int* in_sizes, int n_in,
                              void* d_out, int out_size, void* d_ws, size_t ws_size,
                              hipStream_t stream) {
  const float* images  = (const float*)d_in[0];
  const float* conv1_w = (const float*)d_in[1];
  const float* conv1_b = (const float*)d_in[2];
  const float* conv2_w = (const float*)d_in[3];
  const float* conv2_b = (const float*)d_in[4];
  const float* fc1_w   = (const float*)d_in[5];
  const float* fc1_b   = (const float*)d_in[6];
  const float* fc2_w   = (const float*)d_in[7];
  const float* fc2_b   = (const float*)d_in[8];
  const float* fc3_w   = (const float*)d_in[9];
  const float* fc3_b   = (const float*)d_in[10];
  const float* fcf_w   = (const float*)d_in[11];
  const float* fcf_b   = (const float*)d_in[12];

  char* ws = (char*)d_ws;
  size_t off = 0;
  auto take = [&](size_t bytes) -> void* {
    void* p = ws + off;
    off = (off + bytes + 255) & ~(size_t)255;
    return p;
  };
  _Float16* w1h    = (_Float16*)take((size_t)HID * FLAT_K * 2);
  _Float16* w2h    = (_Float16*)take((size_t)HID * HID * 2);
  _Float16* w3h    = (_Float16*)take((size_t)HID * HID * 2);
  _Float16* wfh    = (_Float16*)take((size_t)32 * HID * 2);
  _Float16* w1colg = (_Float16*)take((size_t)32 * 32 * 2);
  _Float16* b2colg = (_Float16*)take((size_t)32 * C2K * 2);
  _Float16* dump   = (_Float16*)take(256);
  _Float16* act0   = (_Float16*)take((size_t)NT_TOT * FLAT_K * 2);

  convert_weights_kernel<<<(HID * FLAT_K + 255) / 256, 256, 0, stream>>>(
      fc1_w, fc2_w, fc3_w, fcf_w, conv1_w, conv2_w,
      w1h, w2h, w3h, wfh, w1colg, b2colg);

  conv_tiles_kernel<<<NT_TOT / 4, 128, 0, stream>>>(
      images, w1colg, conv1_b, b2colg, conv2_b, act0, dump);

  mlp_kernel<<<MT16, 128, 0, stream>>>(
      act0, w1h, w2h, w3h, wfh,
      fc1_b, fc2_b, fc3_b, fcf_b, (float*)d_out);
}
